// Attention_36490042147138
// MI455X (gfx1250) — matrix operand order repaired and checked by reading, not compiled, not run on hardware
//
#include <hip/hip_runtime.h>
#include <hip/hip_bf16.h>

#define BATCH  2
#define SEQ    2048
#define DMODEL 1024
#define NHEAD  16
#define DHEAD  64
#define BH     (BATCH*NHEAD)        // 32 head-instances

typedef __bf16 bf16_t;
typedef __attribute__((ext_vector_type(16))) __bf16 v16bf;
typedef __attribute__((ext_vector_type(8)))  __bf16 v8bf;
typedef __attribute__((ext_vector_type(4)))  __bf16 v4bf;
typedef __attribute__((ext_vector_type(8)))  float  v8f;

// ---- WMMA helpers -----------------------------------------------------------
// v_wmma_f32_16x16x32_bf16: D = A(16x32) * B(32x16) + C(16x16 f32)
__device__ __forceinline__ v8f wmma_bf16(v16bf a, v16bf b, v8f c) {
  return __builtin_amdgcn_wmma_f32_16x16x32_bf16(
      /*neg_a=*/false, a, /*neg_b=*/false, b,
      /*c_mod=*/(short)0, c, /*reuse_a=*/false, /*reuse_b=*/false);
}

// A-matrix per-lane layout (ISA 7.12.2, 16-bit A 16x32): lane m=lid&15,
// hi=lid>>4; VGPRs 0-3 = row m, K in [8*hi, 8*hi+8); VGPRs 4-7 = K in
// [16+8*hi, +8). rowp points at 32 contiguous bf16 of row m's K-slice.
__device__ __forceinline__ v16bf make_afrag(const bf16_t* rowp, int hi) {
  v8bf lo = *(const v8bf*)(rowp + 8*hi);
  v8bf hh = *(const v8bf*)(rowp + 16 + 8*hi);
  return __builtin_shufflevector(lo, hh, 0,1,2,3,4,5,6,7,8,9,10,11,12,13,14,15);
}

// B-matrix per-lane layout: lane n=lid&15, hi=lid>>4; VGPRs 0-3 = K in
// [8*hi, 8*hi+8), VGPRs 4-7 = K in [16+8*hi, +8) of column n.  colp points at B^T row n (32 contiguous K).
__device__ __forceinline__ v16bf make_bfrag(const bf16_t* colp, int hi) {
  v8bf lo = *(const v8bf*)(colp + 8*hi);
  v8bf hh = *(const v8bf*)(colp + 16 + 8*hi);
  return __builtin_shufflevector(lo, hh, 0,1,2,3,4,5,6,7,8,9,10,11,12,13,14,15);
}

__device__ __forceinline__ v4bf cvt4(float4 f) {
  v4bf r; r[0]=(bf16_t)f.x; r[1]=(bf16_t)f.y; r[2]=(bf16_t)f.z; r[3]=(bf16_t)f.w;
  return r;
}

// ============================================================================
// Kernel 1: fused QKV projection  H = X @ Wq + bq  (same Wq for q,k,v),
// output head-split bf16:  H[b][h][l][d].
// M = 3*B*L = 12288 (grid.x=96 tiles of 128), N = 1024 (grid.y=8 tiles of 128),
// K = 1024 in steps of 32.  8 waves: 4 along M (32 rows) x 2 along N (64 cols).
// ============================================================================
__global__ __launch_bounds__(256)
void proj_kernel(const float* __restrict__ q, const float* __restrict__ k,
                 const float* __restrict__ v, const float* __restrict__ Wq,
                 const float* __restrict__ bq,
                 bf16_t* __restrict__ qh, bf16_t* __restrict__ kh,
                 bf16_t* __restrict__ vh)
{
  __shared__ bf16_t ldsA[128*40];   // X tile, row-major [128][32] (+8 pad)
  __shared__ bf16_t ldsB[128*40];   // Wq tile transposed: [n][k]

  const int t    = threadIdx.x;
  const int mblk = blockIdx.x;                 // 0..95
  const int n0   = blockIdx.y * 128;
  const int xsel = mblk >> 5;                  // 4096 rows per input, 32 blocks
  const int r0   = (mblk & 31) * 128;          // row within this input
  const float* X = (xsel == 0) ? q : (xsel == 1) ? k : v;
  bf16_t*      H = (xsel == 0) ? qh : (xsel == 1) ? kh : vh;

  const int w = t >> 5, lid = t & 31, lane16 = lid & 15, hi = lid >> 4;
  const int waveM = w & 3;       // 32-row stripe
  const int waveN = w >> 2;      // 64-col stripe

  v8f acc[2][4];
  #pragma unroll
  for (int mt = 0; mt < 2; ++mt)
    #pragma unroll
    for (int nt = 0; nt < 4; ++nt) acc[mt][nt] = (v8f){0,0,0,0,0,0,0,0};

  const int arow = t >> 1;              // A loader: 2 threads / row
  const int acb  = (t & 1) * 16;
  const int bk   = t >> 3;              // B loader: k row 0..31
  const int bnb  = (t & 7) * 16;        // n base

  for (int k0 = 0; k0 < DMODEL; k0 += 32) {
    __syncthreads();
    { // A tile: X[r0+arow][k0+acb .. +16) -> bf16
      const float* src = X + (size_t)(r0 + arow) * DMODEL + k0 + acb;
      float4 f0 = *(const float4*)(src + 0);
      float4 f1 = *(const float4*)(src + 4);
      float4 f2 = *(const float4*)(src + 8);
      float4 f3 = *(const float4*)(src + 12);
      bf16_t* dst = &ldsA[arow*40 + acb];
      *(v4bf*)(dst + 0)  = cvt4(f0);
      *(v4bf*)(dst + 4)  = cvt4(f1);
      *(v4bf*)(dst + 8)  = cvt4(f2);
      *(v4bf*)(dst + 12) = cvt4(f3);
    }
    { // B tile: Wq[k0+bk][n0+bnb .. +16) -> transposed ldsB[n][k]
      const float* src = Wq + (size_t)(k0 + bk) * DMODEL + n0 + bnb;
      float fs[16];
      *(float4*)&fs[0]  = *(const float4*)(src + 0);
      *(float4*)&fs[4]  = *(const float4*)(src + 4);
      *(float4*)&fs[8]  = *(const float4*)(src + 8);
      *(float4*)&fs[12] = *(const float4*)(src + 12);
      #pragma unroll
      for (int j = 0; j < 16; ++j) ldsB[(bnb + j)*40 + bk] = (bf16_t)fs[j];
    }
    __syncthreads();

    v16bf afr[2], bfr[4];
    #pragma unroll
    for (int mt = 0; mt < 2; ++mt)
      afr[mt] = make_afrag(&ldsA[(waveM*32 + mt*16 + lane16)*40], hi);
    #pragma unroll
    for (int nt = 0; nt < 4; ++nt)
      bfr[nt] = make_bfrag(&ldsB[(waveN*64 + nt*16 + lane16)*40], hi);
    #pragma unroll
    for (int mt = 0; mt < 2; ++mt)
      #pragma unroll
      for (int nt = 0; nt < 4; ++nt)
        acc[mt][nt] = wmma_bf16(afr[mt], bfr[nt], acc[mt][nt]);
  }

  // Epilogue: + bias, convert bf16, scatter into head-split layout.
  #pragma unroll
  for (int nt = 0; nt < 4; ++nt) {
    const int gcol = n0 + waveN*64 + nt*16 + lane16;
    const int hd = gcol >> 6, d = gcol & 63;
    const float bias = bq[gcol];
    #pragma unroll
    for (int mt = 0; mt < 2; ++mt) {
      #pragma unroll
      for (int r = 0; r < 8; ++r) {
        const int grow = r0 + waveM*32 + mt*16 + hi*8 + r;   // 0..4095
        const int b = grow >> 11, l = grow & 2047;
        H[((size_t)(b*NHEAD + hd)*SEQ + l)*DHEAD + d] =
            (bf16_t)(acc[mt][nt][r] + bias);
      }
    }
  }
}

// ============================================================================
// Kernel 2: scores = (Q @ K^T) / 8  per head-instance, f32 into d_out attn
// region.  M=N=128 tiles, K=64 (two WMMA k-steps).  B^T tile == Kh rows.
// ============================================================================
__global__ __launch_bounds__(256)
void scores_kernel(const bf16_t* __restrict__ qh, const bf16_t* __restrict__ kh,
                   float* __restrict__ attn)
{
  __shared__ bf16_t ldsQ[128*72];   // [row][64] (+8 pad)
  __shared__ bf16_t ldsK[128*72];

  const int t  = threadIdx.x;
  const int m0 = blockIdx.x * 128;
  const int n0 = blockIdx.y * 128;
  const int bh = blockIdx.z;
  const bf16_t* Q = qh + (size_t)bh * SEQ * DHEAD;
  const bf16_t* K = kh + (size_t)bh * SEQ * DHEAD;

  { // load both 128x64 bf16 tiles (2 threads / row, 32 elems each)
    const int row = t >> 1, cb = (t & 1) * 32;
    const v8bf* qs = (const v8bf*)(Q + (size_t)(m0 + row)*DHEAD + cb);
    const v8bf* ks = (const v8bf*)(K + (size_t)(n0 + row)*DHEAD + cb);
    #pragma unroll
    for (int j = 0; j < 4; ++j) {
      *(v8bf*)&ldsQ[row*72 + cb + j*8] = qs[j];
      *(v8bf*)&ldsK[row*72 + cb + j*8] = ks[j];
    }
  }
  __syncthreads();

  const int w = t >> 5, lid = t & 31, lane16 = lid & 15, hi = lid >> 4;
  const int waveM = w & 3, waveN = w >> 2;

  v8f acc[2][4];
  #pragma unroll
  for (int mt = 0; mt < 2; ++mt)
    #pragma unroll
    for (int nt = 0; nt < 4; ++nt) acc[mt][nt] = (v8f){0,0,0,0,0,0,0,0};

  #pragma unroll
  for (int kk = 0; kk < 2; ++kk) {
    v16bf afr[2], bfr[4];
    #pragma unroll
    for (int mt = 0; mt < 2; ++mt)
      afr[mt] = make_afrag(&ldsQ[(waveM*32 + mt*16 + lane16)*72 + kk*32], hi);
    #pragma unroll
    for (int nt = 0; nt < 4; ++nt)
      bfr[nt] = make_bfrag(&ldsK[(waveN*64 + nt*16 + lane16)*72 + kk*32], hi);
    #pragma unroll
    for (int mt = 0; mt < 2; ++mt)
      #pragma unroll
      for (int nt = 0; nt < 4; ++nt)
        acc[mt][nt] = wmma_bf16(afr[mt], bfr[nt], acc[mt][nt]);
  }

  float* S = attn + (size_t)bh * SEQ * SEQ;
  #pragma unroll
  for (int mt = 0; mt < 2; ++mt)
    #pragma unroll
    for (int nt = 0; nt < 4; ++nt)
      #pragma unroll
      for (int r = 0; r < 8; ++r) {
        const int grow = m0 + waveM*32 + mt*16 + hi*8 + r;
        const int gcol = n0 + waveN*64 + nt*16 + lane16;
        S[(size_t)grow*SEQ + gcol] = acc[mt][nt][r] * 0.125f;  // 1/sqrt(64)
      }
}

// ============================================================================
// Kernel 3: row softmax in place (attn region), with mask term.
// One block per row (65536 rows), 256 threads x 8 elements.
// ============================================================================
__global__ __launch_bounds__(256)
void softmax_kernel(float* __restrict__ attn, const float* __restrict__ mask)
{
  const int t = threadIdx.x;
  const int row = blockIdx.x;              // ((b*16+h)*2048 + l)
  const int b = row >> 15;                 // 32768 rows per batch
  float* rowp = attn + (size_t)row * SEQ;
  const float* mrow = mask + (size_t)b * SEQ;
  const int c0 = t * 8;

  float vals[8];
  float4 x0 = *(const float4*)(rowp + c0);
  float4 x1 = *(const float4*)(rowp + c0 + 4);
  float4 m0 = *(const float4*)(mrow + c0);
  float4 m1 = *(const float4*)(mrow + c0 + 4);
  vals[0] = x0.x + (1.0f - m0.x) * -1e9f;
  vals[1] = x0.y + (1.0f - m0.y) * -1e9f;
  vals[2] = x0.z + (1.0f - m0.z) * -1e9f;
  vals[3] = x0.w + (1.0f - m0.w) * -1e9f;
  vals[4] = x1.x + (1.0f - m1.x) * -1e9f;
  vals[5] = x1.y + (1.0f - m1.y) * -1e9f;
  vals[6] = x1.z + (1.0f - m1.z) * -1e9f;
  vals[7] = x1.w + (1.0f - m1.w) * -1e9f;

  __shared__ float red[8];
  float mx = vals[0];
  #pragma unroll
  for (int i = 1; i < 8; ++i) mx = fmaxf(mx, vals[i]);
  #pragma unroll
  for (int off = 16; off >= 1; off >>= 1) mx = fmaxf(mx, __shfl_xor(mx, off, 32));
  if ((t & 31) == 0) red[t >> 5] = mx;
  __syncthreads();
  mx = red[0];
  #pragma unroll
  for (int i = 1; i < 8; ++i) mx = fmaxf(mx, red[i]);
  __syncthreads();

  float s = 0.0f;
  #pragma unroll
  for (int i = 0; i < 8; ++i) { vals[i] = __expf(vals[i] - mx); s += vals[i]; }
  #pragma unroll
  for (int off = 16; off >= 1; off >>= 1) s += __shfl_xor(s, off, 32);
  if ((t & 31) == 0) red[t >> 5] = s;
  __syncthreads();
  s = red[0];
  #pragma unroll
  for (int i = 1; i < 8; ++i) s += red[i];

  const float inv = 1.0f / s;
  float4 o0 = make_float4(vals[0]*inv, vals[1]*inv, vals[2]*inv, vals[3]*inv);
  float4 o1 = make_float4(vals[4]*inv, vals[5]*inv, vals[6]*inv, vals[7]*inv);
  *(float4*)(rowp + c0)     = o0;
  *(float4*)(rowp + c0 + 4) = o1;
}

// ============================================================================
// Kernel 4: out = attn @ vh per head-instance.  M=2048 (16 blocks of 128),
// N=64, K=2048 in steps of 32.  8 waves, each a 16x64 stripe (1x4 tiles).
// attn loaded f32 -> bf16 on the fly; vh tile transposed into LDS.
// ============================================================================
__global__ __launch_bounds__(256)
void av_kernel(const float* __restrict__ attn, const bf16_t* __restrict__ vh,
               float* __restrict__ out)
{
  __shared__ bf16_t ldsA[128*40];   // attn tile [128][32] (+8 pad)
  __shared__ bf16_t ldsB[64*40];    // vh^T tile [d][k]

  const int t  = threadIdx.x;
  const int m0 = blockIdx.x * 128;
  const int bh = blockIdx.z;
  const float*  A = attn + (size_t)bh * SEQ * SEQ;
  const bf16_t* V = vh   + (size_t)bh * SEQ * DHEAD;

  const int w = t >> 5, lid = t & 31, lane16 = lid & 15, hi = lid >> 4;

  v8f acc[4];
  #pragma unroll
  for (int nt = 0; nt < 4; ++nt) acc[nt] = (v8f){0,0,0,0,0,0,0,0};

  const int arow = t >> 1, acb = (t & 1) * 16;
  const int vk  = t >> 3;            // k row 0..31
  const int vdb = (t & 7) * 8;       // d base 0..56

  for (int k0 = 0; k0 < SEQ; k0 += 32) {
    __syncthreads();
    { // attn tile f32 -> bf16
      const float* src = A + (size_t)(m0 + arow)*SEQ + k0 + acb;
      float4 f0 = *(const float4*)(src + 0);
      float4 f1 = *(const float4*)(src + 4);
      float4 f2 = *(const float4*)(src + 8);
      float4 f3 = *(const float4*)(src + 12);
      bf16_t* dst = &ldsA[arow*40 + acb];
      *(v4bf*)(dst + 0)  = cvt4(f0);
      *(v4bf*)(dst + 4)  = cvt4(f1);
      *(v4bf*)(dst + 8)  = cvt4(f2);
      *(v4bf*)(dst + 12) = cvt4(f3);
    }
    { // vh tile, transpose [k][d] -> ldsB[d][k]
      const bf16_t* src = V + (size_t)(k0 + vk)*DHEAD + vdb;
      v8bf d8 = *(const v8bf*)src;
      #pragma unroll
      for (int j = 0; j < 8; ++j) ldsB[(vdb + j)*40 + vk] = d8[j];
    }
    __syncthreads();

    v16bf afr = make_afrag(&ldsA[(w*16 + lane16)*40], hi);
    #pragma unroll
    for (int nt = 0; nt < 4; ++nt) {
      v16bf bfr = make_bfrag(&ldsB[(nt*16 + lane16)*40], hi);
      acc[nt] = wmma_bf16(afr, bfr, acc[nt]);
    }
  }

  float* O = out + (size_t)bh * SEQ * DHEAD;
  #pragma unroll
  for (int nt = 0; nt < 4; ++nt)
    #pragma unroll
    for (int r = 0; r < 8; ++r) {
      const int grow = m0 + w*16 + hi*8 + r;
      const int gcol = nt*16 + lane16;
      O[(size_t)grow*DHEAD + gcol] = acc[nt][r];
    }
}

// ============================================================================
extern "C" void kernel_launch(void* const* d_in, const int* in_sizes, int n_in,
                              void* d_out, int out_size, void* d_ws, size_t ws_size,
                              hipStream_t stream)
{
  const float* q    = (const float*)d_in[0];
  const float* k    = (const float*)d_in[1];
  const float* v    = (const float*)d_in[2];
  const float* mask = (const float*)d_in[3];
  const float* Wq   = (const float*)d_in[4];
  const float* bq   = (const float*)d_in[5];

  const size_t HSZ = (size_t)BATCH * NHEAD * SEQ * DHEAD;   // 4,194,304
  float* out  = (float*)d_out;
  float* attn = out + HSZ;                                   // tuple order: (out, attn)

  bf16_t* qh = (bf16_t*)d_ws;          // 8 MB each, 24 MB total
  bf16_t* kh = qh + HSZ;
  bf16_t* vh = kh + HSZ;

  proj_kernel   <<<dim3(96, 8, 1),        256, 0, stream>>>(q, k, v, Wq, bq, qh, kh, vh);
  scores_kernel <<<dim3(16, 16, BH),      256, 0, stream>>>(qh, kh, attn);
  softmax_kernel<<<dim3(BATCH*NHEAD*SEQ), 256, 0, stream>>>(attn, mask);
  av_kernel     <<<dim3(16, 1, BH),       256, 0, stream>>>(attn, vh, out);
}
